// UniPhyBlock_44281112822253
// MI455X (gfx1250) — matrix operand order repaired and checked by reading, not compiled, not run on hardware
//
#include <hip/hip_runtime.h>
#include <hip/hip_fp16.h>
#include <stdint.h>

// ---------------------------------------------------------------------------
// UniPhyBlock on gfx1250 (MI455X).
// All matmuls -> v_wmma_f32_16x16x32_f16 (block tile 128x64, 8 waves/WG).
// Tile staging: TDM (tensor_load_to_lds) for B tiles, async global->LDS
// (global_load_async_to_lds_b128) for A tiles, with sync fallbacks.
// B operands pre-transposed to [N][K] so every staging access is b128.
// ---------------------------------------------------------------------------

#define BT_   32
#define HW_   4096
#define MROWS 131072   // B*T*H*W rows
#define KCONV 2304

typedef __attribute__((ext_vector_type(16))) _Float16 v16h;
typedef __attribute__((ext_vector_type(8)))  float    v8f;
typedef __attribute__((ext_vector_type(4)))  float    v4f;
typedef __attribute__((ext_vector_type(4)))  unsigned int u32x4;
typedef __attribute__((ext_vector_type(8)))  int      i32x8;
typedef __attribute__((ext_vector_type(4)))  int      i32x4;

typedef __attribute__((address_space(1))) i32x4 gi32x4;   // global, b128-typed
typedef __attribute__((address_space(3))) i32x4 li32x4;   // LDS, b128-typed
typedef __attribute__((address_space(3))) void  as3_void;

#if __has_builtin(__builtin_amdgcn_global_load_async_to_lds_b128)
#define HAVE_ASYNC_LDS 1
#else
#define HAVE_ASYNC_LDS 0
#endif
#if __has_builtin(__builtin_amdgcn_tensor_load_to_lds)
#define HAVE_TDM 1
#else
#define HAVE_TDM 0
#endif

union HFrag { v4f q[2]; v16h h; };

__device__ __forceinline__ unsigned lds_offset_of(const void* p) {
  return (unsigned)(unsigned long long)(as3_void*)p;
}

#if HAVE_ASYNC_LDS
__device__ __forceinline__ void async_b128(const _Float16* g, _Float16* l) {
  __builtin_amdgcn_global_load_async_to_lds_b128((gi32x4*)g, (li32x4*)l, 0, 0);
}
__device__ __forceinline__ void wait_async0() {
#if __has_builtin(__builtin_amdgcn_s_wait_asynccnt)
  __builtin_amdgcn_s_wait_asynccnt(0);
#else
  asm volatile("s_wait_asynccnt 0x0" ::: "memory");
#endif
}
#endif

#if HAVE_TDM
// Issue one TDM 2D tile load: `rows` rows of `row_elems` f16, global row
// stride `row_stride_elems`, packed contiguously at LDS byte offset lds_off.
// D# layout per CDNA5 ISA 8.3/8.4 (group0: lds/global addr + type=2;
// group1: data_size=2B, tensor dims, tile dims, dim0 stride).
__device__ __forceinline__ void tdm_load_2d_to_lds(
    const _Float16* gsrc, unsigned lds_off,
    unsigned rows, unsigned row_elems, unsigned long long row_stride_elems)
{
  unsigned long long ga = (unsigned long long)(uintptr_t)gsrc;
  u32x4 g0;
  g0[0] = 1u;                                                  // count=1
  g0[1] = lds_off;                                             // lds_addr
  g0[2] = (unsigned)ga;                                        // global_addr lo
  g0[3] = (unsigned)((ga >> 32) & 0x01FFFFFFull) | (2u << 30); // hi | type=2
  i32x8 g1;
  unsigned td0 = 0x7FFFFFFFu;                                  // tensor_dim0 (no clip)
  unsigned td1 = rows;                                         // tensor_dim1
  g1[0] = (int)(1u << 16);                                     // data_size=1 (2B)
  g1[1] = (int)((td0 & 0xFFFFu) << 16);                        // td0[15:0]
  g1[2] = (int)(((td0 >> 16) & 0xFFFFu) | ((td1 & 0xFFFFu) << 16));
  g1[3] = (int)(((td1 >> 16) & 0xFFFFu) | ((row_elems & 0xFFFFu) << 16)); // tile_dim0
  g1[4] = (int)(rows & 0xFFFFu);                               // tile_dim1
  g1[5] = (int)(row_stride_elems & 0xFFFFFFFFull);             // dim0 stride lo
  g1[6] = (int)((row_stride_elems >> 32) & 0xFFFFull);         // dim0 stride hi
  g1[7] = 0;
  i32x4 g2; g2[0] = 0; g2[1] = 0; g2[2] = 0; g2[3] = 0;
  i32x4 g3 = g2;
#if __clang_major__ >= 23
  i32x8 g4; g4[0]=0; g4[1]=0; g4[2]=0; g4[3]=0; g4[4]=0; g4[5]=0; g4[6]=0; g4[7]=0;
  __builtin_amdgcn_tensor_load_to_lds(g0, g1, g2, g3, g4, 0);
#else
  __builtin_amdgcn_tensor_load_to_lds(g0, g1, g2, g3, 0);
#endif
}
#endif

// ---- WMMA fragment loaders (wave32 layouts per CDNA5 ISA 7.12.2) ----------
__device__ __forceinline__ v16h load_frag_a(const _Float16* As, int rowBase, int lane) {
  int m  = lane & 15;
  int kb = (lane >> 4) << 3;              // 0 or 8
  const _Float16* p = As + (rowBase + m) * 32;
  HFrag f;
  f.q[0] = *(const v4f*)(p + kb);
  f.q[1] = *(const v4f*)(p + kb + 16);
  return f.h;
}
__device__ __forceinline__ v16h load_frag_b(const _Float16* Bs, int colBase, int lane) {
  int n  = lane & 15;
  int kb = (lane >> 4) << 3;              // 0 or 8
  const _Float16* p = Bs + (colBase + n) * 32 + kb;
  HFrag f;
  f.q[0] = *(const v4f*)(p);
  f.q[1] = *(const v4f*)(p + 16);
  return f.h;
}

// ---------------------------------------------------------------------------
// Generic f16 GEMM: Out[M,N] = s_ab*(A[M,K] @ Bt^T) + s_c*Cin, Bt is [N][K].
// ---------------------------------------------------------------------------
__global__ __launch_bounds__(256) void gemm_f16_kernel(
    const _Float16* __restrict__ A, const _Float16* __restrict__ Bt,
    const float* __restrict__ Cin, float* __restrict__ Out,
    int M, int N, int Kd, float s_ab, float s_c)
{
  __shared__ _Float16 As[128 * 32];
  __shared__ _Float16 Bs[64 * 32];

  const int tid   = threadIdx.x;
  const int lane  = tid & 31;
  const int wv    = tid >> 5;
  const int waveM = wv & 3;
  const int waveN = wv >> 2;
  const long blockM = (long)blockIdx.y * 128;
  const long blockN = (long)blockIdx.x * 64;

  v8f acc00 = {}; v8f acc01 = {}; v8f acc10 = {}; v8f acc11 = {};
#if HAVE_TDM
  const unsigned bsOff = lds_offset_of(Bs);
#endif

  for (int k0 = 0; k0 < Kd; k0 += 32) {
#if HAVE_TDM
    if (wv == 0)
      tdm_load_2d_to_lds(Bt + blockN * (long)Kd + k0, bsOff, 64, 32,
                         (unsigned long long)Kd);
#endif
    // stage A: 128x32 halves, 16 contiguous per thread
    {
      int r = tid >> 1;
      int c = (tid & 1) << 4;
      const _Float16* src = A + (blockM + r) * (long)Kd + k0 + c;
      _Float16* dst = As + r * 32 + c;
      __builtin_prefetch((const void*)(src + 32), 0, 1);
#if HAVE_ASYNC_LDS
      async_b128(src, dst);
      async_b128(src + 8, dst + 8);
#else
      *(v4f*)dst       = *(const v4f*)src;
      *(v4f*)(dst + 8) = *(const v4f*)(src + 8);
#endif
    }
#if !HAVE_TDM
    // stage B: 64x32 halves from transposed Bt, 8 contiguous per thread
    {
      int n  = tid >> 2;
      int ks = (tid & 3) << 3;
      const _Float16* src = Bt + (blockN + n) * (long)Kd + k0 + ks;
      _Float16* dst = Bs + n * 32 + ks;
#if HAVE_ASYNC_LDS
      async_b128(src, dst);
#else
      *(v4f*)dst = *(const v4f*)src;
#endif
    }
#endif
#if HAVE_ASYNC_LDS
    wait_async0();
#endif
#if HAVE_TDM
    if (wv == 0) __builtin_amdgcn_s_wait_tensorcnt(0);
#endif
    __syncthreads();

    v16h a0 = load_frag_a(As, waveM * 32,      lane);
    v16h a1 = load_frag_a(As, waveM * 32 + 16, lane);
    v16h b0 = load_frag_b(Bs, waveN * 32,      lane);
    v16h b1 = load_frag_b(Bs, waveN * 32 + 16, lane);

    acc00 = __builtin_amdgcn_wmma_f32_16x16x32_f16(false, a0, false, b0, (short)0, acc00, false, false);
    acc01 = __builtin_amdgcn_wmma_f32_16x16x32_f16(false, a0, false, b1, (short)0, acc01, false, false);
    acc10 = __builtin_amdgcn_wmma_f32_16x16x32_f16(false, a1, false, b0, (short)0, acc10, false, false);
    acc11 = __builtin_amdgcn_wmma_f32_16x16x32_f16(false, a1, false, b1, (short)0, acc11, false, false);
    __syncthreads();
  }

  const int mOff = ((lane >> 4) << 3);
  const int nOff = (lane & 15);
#pragma unroll
  for (int mt = 0; mt < 2; ++mt) {
#pragma unroll
    for (int nt = 0; nt < 2; ++nt) {
      v8f acc = (mt == 0) ? (nt == 0 ? acc00 : acc01) : (nt == 0 ? acc10 : acc11);
#pragma unroll
      for (int i = 0; i < 8; ++i) {
        int  m   = waveM * 32 + mt * 16 + mOff + i;
        int  n   = waveN * 32 + nt * 16 + nOff;
        long idx = (blockM + m) * (long)N + blockN + n;
        float base = Cin ? Cin[idx] : 0.f;
        Out[idx] = s_ab * acc[i] + s_c * base;
      }
    }
  }
}

// ---------------------------------------------------------------------------
// Conv 3x3 as implicit-im2col WMMA GEMM; B = wBt [256][2304] (already [N][K]).
// ---------------------------------------------------------------------------
__global__ __launch_bounds__(256) void conv_gemm_kernel(
    const _Float16* __restrict__ xri,   // [BT,256,64,64]
    const _Float16* __restrict__ wBt,   // [256][2304]
    const float*    __restrict__ bias,  // [256]
    float*          __restrict__ Out)   // [MROWS,256]
{
  __shared__ _Float16 As[128 * 32];
  __shared__ _Float16 Bs[64 * 32];

  const int tid   = threadIdx.x;
  const int lane  = tid & 31;
  const int wv    = tid >> 5;
  const int waveM = wv & 3;
  const int waveN = wv >> 2;
  const long blockM = (long)blockIdx.y * 128;
  const long blockN = (long)blockIdx.x * 64;

  v8f acc00 = {}; v8f acc01 = {}; v8f acc10 = {}; v8f acc11 = {};
#if HAVE_TDM
  const unsigned bsOff = lds_offset_of(Bs);
#endif

  const int rA  = tid >> 1;
  const int c0A = (tid & 1) << 4;
  const long mRow = blockM + rA;
  const int bt = (int)(mRow >> 12);
  const int hw = (int)(mRow & 4095);
  const int hh = hw >> 6;
  const int ww = hw & 63;

  for (int k0 = 0; k0 < KCONV; k0 += 32) {
#if HAVE_TDM
    if (wv == 0)
      tdm_load_2d_to_lds(wBt + blockN * (long)KCONV + k0, bsOff, 64, 32,
                         (unsigned long long)KCONV);
#endif
    // stage A with on-the-fly im2col (zero padding at borders)
#pragma unroll 4
    for (int i = 0; i < 16; ++i) {
      int kk = c0A + i;
      int k  = k0 + kk;
      int ci = k / 9;
      int r9 = k - ci * 9;
      int kh = r9 / 3;
      int kw = r9 - kh * 3;
      int ih = hh + kh - 1;
      int iw = ww + kw - 1;
      _Float16 v = (_Float16)0.f;
      if ((unsigned)ih < 64u && (unsigned)iw < 64u)
        v = xri[(((long)bt * 256 + ci) << 12) + (ih << 6) + iw];
      As[rA * 32 + kk] = v;
    }
#if !HAVE_TDM
    {
      int n  = tid >> 2;
      int ks = (tid & 3) << 3;
      const _Float16* src = wBt + (blockN + n) * (long)KCONV + k0 + ks;
      _Float16* dst = Bs + n * 32 + ks;
#if HAVE_ASYNC_LDS
      async_b128(src, dst);
#else
      *(v4f*)dst = *(const v4f*)src;
#endif
    }
#if HAVE_ASYNC_LDS
    wait_async0();
#endif
#endif
#if HAVE_TDM
    if (wv == 0) __builtin_amdgcn_s_wait_tensorcnt(0);
#endif
    __syncthreads();

    v16h a0 = load_frag_a(As, waveM * 32,      lane);
    v16h a1 = load_frag_a(As, waveM * 32 + 16, lane);
    v16h b0 = load_frag_b(Bs, waveN * 32,      lane);
    v16h b1 = load_frag_b(Bs, waveN * 32 + 16, lane);

    acc00 = __builtin_amdgcn_wmma_f32_16x16x32_f16(false, a0, false, b0, (short)0, acc00, false, false);
    acc01 = __builtin_amdgcn_wmma_f32_16x16x32_f16(false, a0, false, b1, (short)0, acc01, false, false);
    acc10 = __builtin_amdgcn_wmma_f32_16x16x32_f16(false, a1, false, b0, (short)0, acc10, false, false);
    acc11 = __builtin_amdgcn_wmma_f32_16x16x32_f16(false, a1, false, b1, (short)0, acc11, false, false);
    __syncthreads();
  }

  const int mOff = ((lane >> 4) << 3);
  const int nOff = (lane & 15);
#pragma unroll
  for (int mt = 0; mt < 2; ++mt) {
#pragma unroll
    for (int nt = 0; nt < 2; ++nt) {
      v8f acc = (mt == 0) ? (nt == 0 ? acc00 : acc01) : (nt == 0 ? acc10 : acc11);
#pragma unroll
      for (int i = 0; i < 8; ++i) {
        int  m = waveM * 32 + mt * 16 + mOff + i;
        int  n = waveN * 32 + nt * 16 + nOff;
        long idx = (blockM + m) * 256L + blockN + n;
        Out[idx] = acc[i] + bias[blockN + n];
      }
    }
  }
}

// ---------------------------------------------------------------------------
// Elementwise / LN / scan kernels
// ---------------------------------------------------------------------------
__global__ void f32_to_f16_kernel(const float* __restrict__ src,
                                  _Float16* __restrict__ dst, long n) {
  long i = (long)blockIdx.x * 256 + threadIdx.x;
  if (i < n) dst[i] = (_Float16)src[i];
}

// dst[N][K] = src[K][N], f32 -> f16
__global__ void transpose_f32_to_f16_kernel(const float* __restrict__ src,
                                            _Float16* __restrict__ dst, int K, int N) {
  long i = (long)blockIdx.x * 256 + threadIdx.x;
  if (i >= (long)K * N) return;
  int k = (int)(i / N);
  int n = (int)(i - (long)k * N);
  dst[(long)n * K + k] = (_Float16)src[i];
}

__global__ __launch_bounds__(256) void ln_spatial_kernel(
    const float* __restrict__ xre, const float* __restrict__ xim,
    const float* __restrict__ g, const float* __restrict__ bb,
    const float* __restrict__ metric, _Float16* __restrict__ xri)
{
  __shared__ float red[256];
  int p  = blockIdx.x;
  int bt = p >> 12;
  int hw = p & 4095;
  int c  = threadIdx.x;
  float v = (c < 128) ? xre[(((long)bt * 128 + c) << 12) + hw]
                      : xim[(((long)bt * 128 + (c - 128)) << 12) + hw];
  red[c] = v; __syncthreads();
  for (int s = 128; s > 0; s >>= 1) { if (c < s) red[c] += red[c + s]; __syncthreads(); }
  float mean = red[0] * (1.f / 256.f);
  __syncthreads();
  float d0 = v - mean;
  red[c] = d0 * d0; __syncthreads();
  for (int s = 128; s > 0; s >>= 1) { if (c < s) red[c] += red[c + s]; __syncthreads(); }
  float var = red[0] * (1.f / 256.f);
  float xn = d0 * rsqrtf(var + 1e-5f) * g[c] + bb[c];
  xri[(((long)bt * 256 + c) << 12) + hw] = (_Float16)(xn * metric[((long)c << 12) + hw]);
}

__global__ void post_conv_kernel(const float* __restrict__ conv_out,
                                 const float* __restrict__ xre,
                                 const float* __restrict__ xim,
                                 float* __restrict__ x1re, float* __restrict__ x1im)
{
  long i = (long)blockIdx.x * 256 + threadIdx.x;
  long m = i >> 7;
  int  d = (int)(i & 127);
  int  bt = (int)(m >> 12);
  int  hw = (int)(m & 4095);
  long xi = (((long)bt * 128 + d) << 12) + hw;
  x1re[xi] = conv_out[m * 256 + d]       + xre[xi];
  x1im[xi] = conv_out[m * 256 + 128 + d] + xim[xi];
}

__global__ __launch_bounds__(256) void ln_temporal_kernel(
    const float* __restrict__ x1re, const float* __restrict__ x1im,
    const float* __restrict__ g, const float* __restrict__ bb,
    _Float16* __restrict__ xtre, _Float16* __restrict__ xtim)
{
  __shared__ float red[256];
  int p  = blockIdx.x;
  int t  = p & 15;
  int n  = p >> 4;
  int b  = n >> 12;
  int hw = n & 4095;
  int c  = threadIdx.x;
  int d  = (c < 128) ? c : (c - 128);
  long xi = (((long)((b * 16 + t) * 128 + d)) << 12) + hw;
  float v = (c < 128) ? x1re[xi] : x1im[xi];
  red[c] = v; __syncthreads();
  for (int s = 128; s > 0; s >>= 1) { if (c < s) red[c] += red[c + s]; __syncthreads(); }
  float mean = red[0] * (1.f / 256.f);
  __syncthreads();
  float d0 = v - mean;
  red[c] = d0 * d0; __syncthreads();
  for (int s = 128; s > 0; s >>= 1) { if (c < s) red[c] += red[c + s]; __syncthreads(); }
  float var = red[0] * (1.f / 256.f);
  float xn = d0 * rsqrtf(var + 1e-5f) * g[c] + bb[c];
  long oi = ((long)p << 7) + d;
  if (c < 128) xtre[oi] = (_Float16)xn; else xtim[oi] = (_Float16)xn;
}

__device__ __forceinline__ unsigned hash_u32(unsigned x) {
  x ^= x >> 16; x *= 0x7FEB352Du; x ^= x >> 15; x *= 0x846CA68Bu; x ^= x >> 16;
  return x;
}

__global__ void scan_kernel(const float* __restrict__ xe_re, const float* __restrict__ xe_im,
                            const float* __restrict__ dt, const float* __restrict__ nu,
                            const float* __restrict__ theta, const float* __restrict__ gain,
                            _Float16* __restrict__ h_re, _Float16* __restrict__ h_im)
{
  long i = (long)blockIdx.x * 256 + threadIdx.x;
  int  d = (int)(i & 127);
  long n = i >> 7;
  int  b = (int)(n >> 12);
  float lr = -log1pf(expf(nu[d]));
  float li = theta[d];
  float gg = gain[d];
  float den = lr * lr + li * li;
  float hr = 0.f, hi = 0.f;
  for (int t = 0; t < 16; ++t) {
    float dte = dt[b * 16 + t];
    float er  = expf(lr * dte);
    float ang = li * dte;
    float dr  = er * cosf(ang), di = er * sinf(ang);
    float nr = dr - 1.f, ni = di;
    float fr = (nr * lr + ni * li) / den;
    float fi = (ni * lr - nr * li) / den;
    long idx = (n * 16 + t) * 128 + d;
    float xr = xe_re[idx], xii = xe_im[idx];
    float ar = xr + tanhf(xr) * gg;
    float ai = xii + tanhf(xii) * gg;
    float ur = ar * fr - ai * fi;
    float ui = ar * fi + ai * fr;
    unsigned hA = hash_u32((unsigned)idx * 0x9E3779B1u + 0x85EBCA6Bu);
    unsigned hB = hash_u32((unsigned)idx * 0xC2B2AE35u + 0x27D4EB2Fu);
    float u1 = (float)(hA & 0xFFFFFF) * (1.f / 16777216.f) + 1e-7f;
    float u2 = (float)(hB & 0xFFFFFF) * (1.f / 16777216.f);
    float rad = sqrtf(-2.f * logf(u1));
    float amp = 0.01f * sqrtf(dte);
    ur += amp * rad * cosf(6.2831853f * u2);
    ui += amp * rad * sinf(6.2831853f * u2);
    float nhr = dr * hr - di * hi + ur;
    float nhi = dr * hi + di * hr + ui;
    hr = nhr; hi = nhi;
    h_re[idx] = (_Float16)hr;
    h_im[idx] = (_Float16)hi;
  }
}

__global__ void post_temporal_kernel(const float* __restrict__ xo_re,
                                     const float* __restrict__ x1re,
                                     const float* __restrict__ x1im,
                                     float* __restrict__ x2re, float* __restrict__ x2im,
                                     _Float16* __restrict__ zre, _Float16* __restrict__ zim)
{
  long i = (long)blockIdx.x * 256 + threadIdx.x;
  int hw = (int)(i & 4095);
  long r = i >> 12;
  int  d = (int)(r & 127);
  int  bt = (int)(r >> 7);
  int  b = bt >> 4;
  int  t = bt & 15;
  long n = ((long)b << 12) + hw;
  float vr = x1re[i] + xo_re[(n * 16 + t) * 128 + d];
  float vi = x1im[i];
  x2re[i] = vr; x2im[i] = vi;
  long mF = ((long)bt << 12) + hw;
  zre[mF * 128 + d] = (_Float16)vr;
  zim[mF * 128 + d] = (_Float16)vi;
}

__global__ void bias_gelu_kernel(const float* __restrict__ hre, const float* __restrict__ him,
                                 const float* __restrict__ b1r, const float* __restrict__ b1i,
                                 _Float16* __restrict__ are, _Float16* __restrict__ aim)
{
  long i = (long)blockIdx.x * 256 + threadIdx.x;
  int col = (int)(i & 511);
  float a = hre[i] + b1r[col];
  float b = him[i] + b1i[col];
  are[i] = (_Float16)(a * 0.5f * (1.f + erff(a * 0.70710678f)));
  aim[i] = (_Float16)(b * 0.5f * (1.f + erff(b * 0.70710678f)));
}

__global__ void final_kernel(const float* __restrict__ x2re, const float* __restrict__ x2im,
                             const float* __restrict__ dre, const float* __restrict__ dim_,
                             const float* __restrict__ b2r, const float* __restrict__ b2i,
                             float* __restrict__ out)
{
  long i = (long)blockIdx.x * 256 + threadIdx.x;
  int hw = (int)(i & 4095);
  long r = i >> 12;
  int  d = (int)(r & 127);
  long bt = r >> 7;
  long mF = (bt << 12) + hw;
  out[i * 2]     = x2re[i] + dre[mF * 128 + d] + b2r[d];
  out[i * 2 + 1] = x2im[i] + dim_[mF * 128 + d] + b2i[d];
}

// Non-launched probe: cluster sync paths
__global__ void cdna5_probe_kernel(int* p) {
  __builtin_amdgcn_s_wait_tensorcnt(0);
  __builtin_amdgcn_s_cluster_barrier();
  if (p) p[0] = __builtin_amdgcn_cluster_id_x();
}

// ---------------------------------------------------------------------------
extern "C" void kernel_launch(void* const* d_in, const int* in_sizes, int n_in,
                              void* d_out, int out_size, void* d_ws, size_t ws_size,
                              hipStream_t stream) {
  (void)in_sizes; (void)n_in; (void)out_size; (void)ws_size;

  const float* x_re   = (const float*)d_in[0];
  const float* x_im   = (const float*)d_in[1];
  const float* dt     = (const float*)d_in[2];
  const float* ln_s_g = (const float*)d_in[3];
  const float* ln_s_b = (const float*)d_in[4];
  const float* ln_t_g = (const float*)d_in[5];
  const float* ln_t_b = (const float*)d_in[6];
  const float* metric = (const float*)d_in[7];
  const float* conv_w = (const float*)d_in[8];
  const float* conv_b = (const float*)d_in[9];
  const float* nu     = (const float*)d_in[10];
  const float* theta  = (const float*)d_in[11];
  const float* enc_re = (const float*)d_in[12];
  const float* enc_im = (const float*)d_in[13];
  const float* dec_re = (const float*)d_in[14];
  const float* dec_im = (const float*)d_in[15];
  const float* s_gain = (const float*)d_in[16];
  const float* w1_re  = (const float*)d_in[17];
  const float* w1_im  = (const float*)d_in[18];
  const float* b1_re  = (const float*)d_in[19];
  const float* b1_im  = (const float*)d_in[20];
  const float* w2_re  = (const float*)d_in[21];
  const float* w2_im  = (const float*)d_in[22];
  const float* b2_re  = (const float*)d_in[23];
  const float* b2_im  = (const float*)d_in[24];
  float* out = (float*)d_out;

  char* ws = (char*)d_ws;
  size_t off = 0;
  auto alloc = [&](size_t bytes) -> char* {
    char* p = ws + off;
    off = (off + bytes + 255) & ~(size_t)255;
    return p;
  };
  const long MD  = MROWS * 128L;
  const long MFF = MROWS * 512L;

  _Float16* xri  = (_Float16*)alloc((size_t)BT_ * 256 * HW_ * 2);
  _Float16* wBt  = (_Float16*)alloc((size_t)256 * KCONV * 2);   // [256][2304]
  _Float16* Ert  = (_Float16*)alloc(128 * 128 * 2);             // transposed [N][K]
  _Float16* Eit  = (_Float16*)alloc(128 * 128 * 2);
  _Float16* Drt  = (_Float16*)alloc(128 * 128 * 2);
  _Float16* Dit  = (_Float16*)alloc(128 * 128 * 2);
  _Float16* W1rt = (_Float16*)alloc(512 * 128 * 2);             // [512][128]
  _Float16* W1it = (_Float16*)alloc(512 * 128 * 2);
  _Float16* W2rt = (_Float16*)alloc(128 * 512 * 2);             // [128][512]
  _Float16* W2it = (_Float16*)alloc(128 * 512 * 2);
  float*    cvo  = (float*)alloc((size_t)MROWS * 256 * 4);
  float*    x1re = (float*)alloc((size_t)MD * 4);
  float*    x1im = (float*)alloc((size_t)MD * 4);
  _Float16* xtre = (_Float16*)alloc((size_t)MD * 2);
  _Float16* xtim = (_Float16*)alloc((size_t)MD * 2);
  float*    xere = (float*)alloc((size_t)MD * 4);
  float*    xeim = (float*)alloc((size_t)MD * 4);
  _Float16* hre  = (_Float16*)alloc((size_t)MD * 2);
  _Float16* him  = (_Float16*)alloc((size_t)MD * 2);
  float*    xore = (float*)alloc((size_t)MD * 4);
  float*    x2re = (float*)alloc((size_t)MD * 4);
  float*    x2im = (float*)alloc((size_t)MD * 4);
  _Float16* zre  = (_Float16*)alloc((size_t)MD * 2);
  _Float16* zim  = (_Float16*)alloc((size_t)MD * 2);
  float*    hdre = (float*)alloc((size_t)MFF * 4);
  float*    hdim = (float*)alloc((size_t)MFF * 4);
  _Float16* acre = (_Float16*)alloc((size_t)MFF * 2);
  _Float16* acim = (_Float16*)alloc((size_t)MFF * 2);
  float*    dre  = (float*)alloc((size_t)MD * 4);
  float*    dim_ = (float*)alloc((size_t)MD * 4);

  const dim3 blk(256);
  auto gblocks = [](long n) { return dim3((unsigned)((n + 255) / 256)); };

  auto gemm = [&](const _Float16* A, const _Float16* Bm, const float* Cin, float* O,
                  int M, int N, int Kd, float sab, float sc) {
    dim3 g((unsigned)(N / 64), (unsigned)(M / 128));
    gemm_f16_kernel<<<g, blk, 0, stream>>>(A, Bm, Cin, O, M, N, Kd, sab, sc);
  };

  // 1) weight prep: transpose all B operands to [N][K] f16
  transpose_f32_to_f16_kernel<<<gblocks(128 * 128), blk, 0, stream>>>(enc_re, Ert, 128, 128);
  transpose_f32_to_f16_kernel<<<gblocks(128 * 128), blk, 0, stream>>>(enc_im, Eit, 128, 128);
  transpose_f32_to_f16_kernel<<<gblocks(128 * 128), blk, 0, stream>>>(dec_re, Drt, 128, 128);
  transpose_f32_to_f16_kernel<<<gblocks(128 * 128), blk, 0, stream>>>(dec_im, Dit, 128, 128);
  transpose_f32_to_f16_kernel<<<gblocks(128 * 512), blk, 0, stream>>>(w1_re, W1rt, 128, 512);
  transpose_f32_to_f16_kernel<<<gblocks(128 * 512), blk, 0, stream>>>(w1_im, W1it, 128, 512);
  transpose_f32_to_f16_kernel<<<gblocks(512 * 128), blk, 0, stream>>>(w2_re, W2rt, 512, 128);
  transpose_f32_to_f16_kernel<<<gblocks(512 * 128), blk, 0, stream>>>(w2_im, W2it, 512, 128);
  // conv_w is [co][ci][kh][kw] == [N][K] already: straight f32->f16 copy
  f32_to_f16_kernel<<<gblocks((long)256 * KCONV), blk, 0, stream>>>(conv_w, wBt, (long)256 * KCONV);

  // 2) spatial LN + metric
  ln_spatial_kernel<<<dim3(MROWS), blk, 0, stream>>>(x_re, x_im, ln_s_g, ln_s_b, metric, xri);

  // 3) conv as WMMA GEMM (+bias)
  conv_gemm_kernel<<<dim3(4, MROWS / 128), blk, 0, stream>>>(xri, wBt, conv_b, cvo);

  // 4) +residual
  post_conv_kernel<<<gblocks(MD), blk, 0, stream>>>(cvo, x_re, x_im, x1re, x1im);

  // 5) temporal LN
  ln_temporal_kernel<<<dim3(MROWS), blk, 0, stream>>>(x1re, x1im, ln_t_g, ln_t_b, xtre, xtim);

  // 6) eigen encode
  gemm(xtre, Ert, nullptr, xere, MROWS, 128, 128,  1.f, 0.f);
  gemm(xtim, Eit, xere,    xere, MROWS, 128, 128, -1.f, 1.f);
  gemm(xtre, Eit, nullptr, xeim, MROWS, 128, 128,  1.f, 0.f);
  gemm(xtim, Ert, xeim,    xeim, MROWS, 128, 128,  1.f, 1.f);

  // 7) forcing + noise + scan
  scan_kernel<<<gblocks((long)8192 * 128), blk, 0, stream>>>(xere, xeim, dt, nu, theta, s_gain, hre, him);

  // 8) eigen decode (real part)
  gemm(hre, Drt, nullptr, xore, MROWS, 128, 128,  1.f, 0.f);
  gemm(him, Dit, xore,    xore, MROWS, 128, 128, -1.f, 1.f);

  // 9) +residual, repack for FFN
  post_temporal_kernel<<<gblocks(MD), blk, 0, stream>>>(xore, x1re, x1im, x2re, x2im, zre, zim);

  // 10) FFN
  gemm(zre, W1rt, nullptr, hdre, MROWS, 512, 128,  1.f, 0.f);
  gemm(zim, W1it, hdre,    hdre, MROWS, 512, 128, -1.f, 1.f);
  gemm(zre, W1it, nullptr, hdim, MROWS, 512, 128,  1.f, 0.f);
  gemm(zim, W1rt, hdim,    hdim, MROWS, 512, 128,  1.f, 1.f);
  bias_gelu_kernel<<<gblocks(MFF), blk, 0, stream>>>(hdre, hdim, b1_re, b1_im, acre, acim);
  gemm(acre, W2rt, nullptr, dre,  MROWS, 128, 512,  1.f, 0.f);
  gemm(acim, W2it, dre,     dre,  MROWS, 128, 512, -1.f, 1.f);
  gemm(acre, W2it, nullptr, dim_, MROWS, 128, 512,  1.f, 0.f);
  gemm(acim, W2rt, dim_,    dim_, MROWS, 128, 512,  1.f, 1.f);

  // 11) final residual + bias
  final_kernel<<<gblocks(MD), blk, 0, stream>>>(x2re, x2im, dre, dim_, b2_re, b2_im, out);
}
